// QuantizeEMA_90787018703332
// MI455X (gfx1250) — compile-verified, matrix operand order checked by reading, not run on hardware
//
#include <hip/hip_runtime.h>

#define D_DIM 256
#define K_DIM 4096
#define N_TOT 16384
#define DECAYF 0.99f
#define EPSF 1e-5f

typedef __attribute__((ext_vector_type(16))) _Float16 v16h;
typedef __attribute__((ext_vector_type(8)))  float    v8f;

union HV16 { _Float16 h[16]; uint4 q[2]; v16h v; };

// ---------------- zero scratch ----------------
__global__ void vq_zero_kernel(float* __restrict__ p, int n) {
  int i = blockIdx.x * 256 + threadIdx.x;
  if (i < n) p[i] = 0.0f;
}

// ---------------- transpose + f16 convert codebook ----------------
// E is [D,K] row-major; produce Et32/Et16 as [K,D] row-major (code vectors contiguous).
__global__ void vq_prep_kernel(const float* __restrict__ E,
                               float* __restrict__ Et32,
                               _Float16* __restrict__ Et16) {
  int tid = blockIdx.x * 256 + threadIdx.x;   // tid = d*K + k
  int k = tid & (K_DIM - 1);
  int d = tid >> 12;
  float v = E[tid];
  Et32[(size_t)k * D_DIM + d] = v;
  Et16[(size_t)k * D_DIM + d] = (_Float16)v;
}

// ---------------- ||e_k||^2 ----------------
__global__ void vq_enorm_kernel(const float* __restrict__ Et32, float* __restrict__ enorm) {
  __shared__ float s[256];
  int k = blockIdx.x, t = threadIdx.x;
  float v = Et32[(size_t)k * D_DIM + t];
  s[t] = v * v; __syncthreads();
  for (int st = 128; st > 0; st >>= 1) { if (t < st) s[t] += s[t + st]; __syncthreads(); }
  if (t == 0) enorm[k] = s[0];
}

// ---------------- WMMA distance matmul + fused argmin ----------------
// Block = 8 waves x 32 = 256 threads, covers 128 rows (wave w -> rows row0..row0+15).
// All waves sweep the same k-tile sequence -> B cachelines shared via WGP$/L2.
// Cross-iteration double buffering: loads of tile kt+1 are in flight while the
// WMMAs of tile kt execute (live ranges overlap -> allocator cannot merge them).
__global__ void __launch_bounds__(256) vq_gemm_argmin_kernel(
    const float* __restrict__ X, const _Float16* __restrict__ Et16,
    const float* __restrict__ enorm, int* __restrict__ idxb,
    float* __restrict__ out_idx_f) {
  const int wave   = threadIdx.x >> 5;
  const int lane   = threadIdx.x & 31;
  const int lane15 = lane & 15;
  const int hi     = lane >> 4;            // which K-half this lane holds
  const int row0   = blockIdx.x * 128 + wave * 16;

  // ---- load A tile (16 rows x 256 D) as f16, resident for entire k sweep ----
  // A 16x32 f16 layout: lane holds row M=lane&15; elements 0..7 = K base..base+7,
  // elements 8..15 = K base+16..base+23, base = i*32 + 8*hi.
  const float* xrow = X + (size_t)(row0 + lane15) * D_DIM;
  HV16 A[8];
#pragma unroll
  for (int i = 0; i < 8; ++i) {
    const int d0 = i * 32 + hi * 8;
    float4 f0 = *(const float4*)(xrow + d0);
    float4 f1 = *(const float4*)(xrow + d0 + 4);
    float4 f2 = *(const float4*)(xrow + d0 + 16);
    float4 f3 = *(const float4*)(xrow + d0 + 20);
    A[i].h[0]  = (_Float16)f0.x; A[i].h[1]  = (_Float16)f0.y;
    A[i].h[2]  = (_Float16)f0.z; A[i].h[3]  = (_Float16)f0.w;
    A[i].h[4]  = (_Float16)f1.x; A[i].h[5]  = (_Float16)f1.y;
    A[i].h[6]  = (_Float16)f1.z; A[i].h[7]  = (_Float16)f1.w;
    A[i].h[8]  = (_Float16)f2.x; A[i].h[9]  = (_Float16)f2.y;
    A[i].h[10] = (_Float16)f2.z; A[i].h[11] = (_Float16)f2.w;
    A[i].h[12] = (_Float16)f3.x; A[i].h[13] = (_Float16)f3.y;
    A[i].h[14] = (_Float16)f3.z; A[i].h[15] = (_Float16)f3.w;
  }

  float vm[8]; int bi[8];
#pragma unroll
  for (int v = 0; v < 8; ++v) { vm[v] = 3.4e38f; bi[v] = 0; }

  // B tile fetch: codebook rows contiguous in Et16 [K,D]; same lane-half K pattern as A.
  auto loadB = [&](HV16* B, int kt) {
    const _Float16* brow = Et16 + (size_t)(kt * 16 + lane15) * D_DIM;
#pragma unroll
    for (int i = 0; i < 8; ++i) {
      const int d0 = i * 32 + hi * 8;
      B[i].q[0] = *(const uint4*)(brow + d0);
      B[i].q[1] = *(const uint4*)(brow + d0 + 16);
    }
  };

  // 8 chained WMMAs + fused score/argmin for one 16x16 output tile.
  auto tile = [&](const HV16* B, int kt) {
    v8f c = {0.f, 0.f, 0.f, 0.f, 0.f, 0.f, 0.f, 0.f};
#pragma unroll
    for (int i = 0; i < 8; ++i) {
      c = __builtin_amdgcn_wmma_f32_16x16x32_f16(false, A[i].v, false, B[i].v,
                                                 (short)0, c, false, false);
    }
    // score = ||e||^2 - 2 x.e  (||x||^2 is constant per row -> irrelevant for argmin)
    const float en  = enorm[kt * 16 + lane15];
    const int  cand = kt * 16 + lane15;
#pragma unroll
    for (int v = 0; v < 8; ++v) {
      float s = fmaf(-2.0f, c[v], en);
      if (s < vm[v]) { vm[v] = s; bi[v] = cand; }   // strict < keeps lowest index
    }
  };

  HV16 B0[8], B1[8];
  loadB(B0, 0);
  for (int kt = 0; kt < K_DIM / 16; kt += 2) {
    loadB(B1, kt + 1);                       // in flight while B0 computes
    __builtin_prefetch(Et16 + (size_t)((kt + 2) * 16 + lane15) * D_DIM, 0, 3);
    tile(B0, kt);
    const int ktn = (kt + 2 < K_DIM / 16) ? kt + 2 : (K_DIM / 16 - 1); // branchless tail
    loadB(B0, ktn);                          // in flight while B1 computes
    tile(B1, kt + 1);
  }

  // C/D layout: c[v] is row M = v + 8*hi, column N = lane15.
  // Butterfly min+argmin over the 16 column lanes (stays inside each half-wave).
#pragma unroll
  for (int off = 1; off < 16; off <<= 1) {
#pragma unroll
    for (int v = 0; v < 8; ++v) {
      float ov = __shfl_xor(vm[v], off, 32);
      int   oi = __shfl_xor(bi[v], off, 32);
      if (ov < vm[v] || (ov == vm[v] && oi < bi[v])) { vm[v] = ov; bi[v] = oi; }
    }
  }
  if (lane15 == 0) {
#pragma unroll
    for (int v = 0; v < 8; ++v) {
      int r = row0 + hi * 8 + v;
      idxb[r] = bi[v];
      out_idx_f[r] = (float)bi[v];
    }
  }
}

// ---------------- gather quantized + loss + scatter segment sums ----------------
__global__ void vq_gather_scatter_kernel(
    const float* __restrict__ X, const float* __restrict__ Et32,
    const int* __restrict__ idxb, float* __restrict__ outq,
    float* __restrict__ St, float* __restrict__ cnt, float* __restrict__ scal) {
  __shared__ float s[256];
  int n = blockIdx.x, t = threadIdx.x;
  int k = idxb[n];
  float x = X[(size_t)n * D_DIM + t];
  float q = Et32[(size_t)k * D_DIM + t];
  outq[(size_t)n * D_DIM + t] = q;        // straight-through == quantized numerically
  float d = q - x;
  atomicAdd(&St[(size_t)k * D_DIM + t], x);
  if (t == 0) atomicAdd(&cnt[k], 1.0f);
  s[t] = d * d; __syncthreads();
  for (int st = 128; st > 0; st >>= 1) { if (t < st) s[t] += s[t + st]; __syncthreads(); }
  if (t == 0) atomicAdd(&scal[0], s[0]);
}

// ---------------- EMA cluster size, debias, n = sum(cs), loss finalize ----------------
__global__ void vq_cs_kernel(const float* __restrict__ ema_cs, const float* __restrict__ cnt,
                             const int* __restrict__ counter, float* __restrict__ out_cs,
                             float* __restrict__ scal, float* __restrict__ out_loss) {
  __shared__ float s[1024];
  int t = threadIdx.x;
  const float om = 1.0f - DECAYF;
  float debias = 1.0f - powf(DECAYF, (float)(counter[0] + 1));
  float local = 0.0f;
  for (int k = t; k < K_DIM; k += 1024) {
    float ncs = ema_cs[k] + (cnt[k] - ema_cs[k]) * om;
    out_cs[k] = ncs;
    local += ncs / debias;
  }
  s[t] = local; __syncthreads();
  for (int st = 512; st > 0; st >>= 1) { if (t < st) s[t] += s[t + st]; __syncthreads(); }
  if (t == 0) {
    scal[1] = s[0];        // n = sum(cs)
    scal[2] = debias;
    out_loss[0] = scal[0] * (1.0f / ((float)N_TOT * (float)D_DIM));
  }
}

// ---------------- new_dw_hidden and new_embeddings ----------------
__global__ void vq_final_kernel(const float* __restrict__ ema_dw, const float* __restrict__ St,
                                const float* __restrict__ new_cs, const float* __restrict__ scal,
                                float* __restrict__ out_emb, float* __restrict__ out_dw) {
  int tid = blockIdx.x * 256 + threadIdx.x;  // tid = d*K + k  ([D,K] row-major outputs)
  int k = tid & (K_DIM - 1);
  int d = tid >> 12;
  const float om = 1.0f - DECAYF;
  float e0  = ema_dw[tid];
  float ndw = e0 + (St[(size_t)k * D_DIM + d] - e0) * om;
  out_dw[tid] = ndw;
  float debias = scal[2], nsum = scal[1];
  float cs = new_cs[k] / debias;
  float csadj = (cs + EPSF) / (nsum + (float)K_DIM * EPSF) * nsum;
  out_emb[tid] = (ndw / debias) / csadj;
}

extern "C" void kernel_launch(void* const* d_in, const int* in_sizes, int n_in,
                              void* d_out, int out_size, void* d_ws, size_t ws_size,
                              hipStream_t stream) {
  (void)in_sizes; (void)n_in; (void)out_size; (void)ws_size;
  const float* X      = (const float*)d_in[0];   // [16,32,32,256] -> [N,D]
  const float* E      = (const float*)d_in[1];   // [D,K]
  const float* ema_cs = (const float*)d_in[2];   // [K]
  const float* ema_dw = (const float*)d_in[3];   // [D,K]
  const int*   ctr    = (const int*)d_in[4];     // scalar

  float* out  = (float*)d_out;
  float* out1 = out;                 // quantized_st      [N*D]      4194304
  float* out2 = out + 4194304;       // e_latent_loss     [1]
  float* out3 = out + 4194305;       // encoding_indices  [N]        16384
  float* out4 = out + 4210689;       // new_embeddings    [D*K]      1048576
  float* out5 = out + 5259265;       // new_cs_hidden     [K]        4096
  float* out6 = out + 5263361;       // new_dw_hidden     [D*K]      1048576

  // workspace partition (~10.6 MB)
  float*     cnt   = (float*)d_ws;                  // K                 (zeroed)
  float*     St    = cnt + K_DIM;                   // K*D  [K,D] layout (zeroed)
  float*     scal  = St + (size_t)K_DIM * D_DIM;    // 64 scalars        (zeroed)
  float*     Et32  = scal + 64;                     // K*D f32, [K,D]
  _Float16*  Et16  = (_Float16*)(Et32 + (size_t)K_DIM * D_DIM);  // K*D f16, [K,D]
  float*     enorm = (float*)(Et16 + (size_t)K_DIM * D_DIM);     // K
  int*       idxb  = (int*)(enorm + K_DIM);                      // N

  const int zn = K_DIM + K_DIM * D_DIM + 64;        // cnt + St + scal contiguous
  vq_zero_kernel<<<(zn + 255) / 256, 256, 0, stream>>>(cnt, zn);
  vq_prep_kernel<<<(D_DIM * K_DIM) / 256, 256, 0, stream>>>(E, Et32, Et16);
  vq_enorm_kernel<<<K_DIM, 256, 0, stream>>>(Et32, enorm);
  vq_gemm_argmin_kernel<<<N_TOT / 128, 256, 0, stream>>>(X, Et16, enorm, idxb, out3);
  vq_gather_scatter_kernel<<<N_TOT, 256, 0, stream>>>(X, Et32, idxb, out1, St, cnt, scal);
  vq_cs_kernel<<<1, 1024, 0, stream>>>(ema_cs, cnt, ctr, out5, scal, out2);
  vq_final_kernel<<<(D_DIM * K_DIM) / 256, 256, 0, stream>>>(ema_dw, St, out5, scal, out4, out6);
}
